// VectorWeightedAttentionIdxVersion_23905787969988
// MI455X (gfx1250) — matrix operand order repaired and checked by reading, not compiled, not run on hardware
//
#include <hip/hip_runtime.h>

// MI455X / gfx1250, wave32. Fused attention + class-weighted AV kernel.
// One workgroup (8 waves) handles one (batch, 16-row tile): QK^T via
// v_wmma_f32_16x16x32_f16, softmax in LDS, attn written once, then the
// weighted AV GEMM with the analytic idx-class decomposition.

#define B_    32
#define H_    512
#define NK_   1024
#define S_    1025
#define JT_   65          // ceil(1040/16) score column tiles
#define JPAD_ 1040        // padded column count (multiple of 16)
#define TINV_ 0.04419417382415922f   // 1 / 22.627416997969522 (= 1/(16*sqrt(2)))

typedef _Float16 v16h __attribute__((ext_vector_type(16)));
typedef _Float16 h8   __attribute__((ext_vector_type(8)));
typedef float    v8f  __attribute__((ext_vector_type(8)));

__device__ __forceinline__ v8f wmma_f16(v16h a, v16h b, v8f c) {
  // (neg_a, A, neg_b, B, c_mod, C, reuse_a, reuse_b)
  return __builtin_amdgcn_wmma_f32_16x16x32_f16(false, a, false, b, (short)0, c,
                                                false, false);
}

__device__ __forceinline__ void cvt4(v16h& d, int o, float4 f) {
  d[o]   = (_Float16)f.x; d[o+1] = (_Float16)f.y;
  d[o+2] = (_Float16)f.z; d[o+3] = (_Float16)f.w;
}

__global__ __launch_bounds__(256)
void VWA_attn_fused_kernel(const float* __restrict__ Qg,
                           const float* __restrict__ Kg,
                           const float* __restrict__ Vg,
                           const float* __restrict__ AW,
                           float* __restrict__ outg,
                           float* __restrict__ attng)
{
  __shared__ alignas(16) _Float16 s_p[16][1048];  // scores -> probs (f16)
  __shared__ alignas(16) _Float16 s_q[16][520];   // q tile (f16)

  const int t     = blockIdx.x;      // row tile 0..64 (64 == query row tile)
  const int b     = blockIdx.y;
  const int tid   = threadIdx.x;
  const int lane  = tid & 31;
  const int wave  = tid >> 5;        // 0..7
  const int n     = lane & 15;       // column within 16x16 fragment
  const int hhalf = lane >> 4;       // 0: lanes 0-15, 1: lanes 16-31
  const int kbA   = hhalf ? 8 : 0;   // A-fragment K interleave base
  const int kbB   = hhalf ? 8 : 0;   // B-fragment K interleave base
  const bool qTile = (t == 64);

  // ---------------- stage Q tile: f32 -> f16 in LDS ----------------
  for (int e = tid; e < 16 * H_; e += 256) {
    const int r = e >> 9, c = e & (H_ - 1);
    int i = 16 * t + r; if (i >= S_) i = S_ - 1;
    s_q[r][c] = (_Float16)Qg[(size_t)(b * S_ + i) * H_ + c];
  }
  __syncthreads();

  // ---------------- phase 1: scores = (Q K^T) / T ----------------
  for (int jt = wave; jt < JT_; jt += 8) {
    const int j0 = jt * 16;
    int j = j0 + n; if (j >= S_) j = S_ - 1;       // clamp (pads fixed later)
    const float* kp = Kg + (size_t)(b * S_ + j) * H_;
    v8f acc = {};
    for (int ch = 0; ch < 16; ++ch) {              // K-dim: 16 chunks of 32
      const int c0 = ch * 32;
      // A fragment (q rows) from LDS, ISA A-interleave
      const h8 lo = *(const h8*)&s_q[n][c0 + kbA];
      const h8 hi = *(const h8*)&s_q[n][c0 + kbA + 16];
      v16h a;
      #pragma unroll
      for (int e = 0; e < 8; ++e) { a[e] = lo[e]; a[8 + e] = hi[e]; }
      // B fragment: two 8-f32 runs (k = kbB.., kbB+16..) of this key row -> f16
      const float4 f0 = *(const float4*)(kp + c0 + kbB);
      const float4 f1 = *(const float4*)(kp + c0 + kbB + 4);
      const float4 f2 = *(const float4*)(kp + c0 + kbB + 16);
      const float4 f3 = *(const float4*)(kp + c0 + kbB + 20);
      v16h bb;
      cvt4(bb, 0, f0); cvt4(bb, 4, f1); cvt4(bb, 8, f2); cvt4(bb, 12, f3);
      acc = wmma_f16(a, bb, acc);
    }
    #pragma unroll
    for (int r = 0; r < 8; ++r) {
      const int M = r + 8 * hhalf;
      s_p[M][j0 + n] = (_Float16)(acc[r] * TINV_);
    }
  }
  __syncthreads();

  // ---------------- phase 2: row softmax in LDS ----------------
  {
    const int row = tid >> 4;
    const int c   = tid & 15;
    for (int j = S_ + c; j < JPAD_; j += 16)       // kill pad columns
      s_p[row][j] = (_Float16)(-65504.0f);
  }
  __syncthreads();
  {
    const int row = tid >> 4;
    const int c   = tid & 15;
    float mx = -3.0e38f;
    for (int j = c; j < JPAD_; j += 16) mx = fmaxf(mx, (float)s_p[row][j]);
    #pragma unroll
    for (int o = 8; o >= 1; o >>= 1) mx = fmaxf(mx, __shfl_xor(mx, o, 16));
    float sm = 0.0f;
    for (int j = c; j < JPAD_; j += 16) {
      const float p = __expf((float)s_p[row][j] - mx);
      s_p[row][j] = (_Float16)p;
      sm += p;
    }
    #pragma unroll
    for (int o = 8; o >= 1; o >>= 1) sm += __shfl_xor(sm, o, 16);
    const float inv = 1.0f / sm;
    const int  i  = 16 * t + row;
    const bool rv = (i < S_);
    for (int j = c; j < JPAD_; j += 16) {
      const float p = (float)s_p[row][j] * inv;
      s_p[row][j] = (_Float16)p;
      if (rv && j < S_) attng[(size_t)(b * S_ + i) * S_ + j] = p;
    }
  }
  __syncthreads();

  // ---------------- phase 3: weighted AV GEMM ----------------
  v8f accF[4] = {{}, {}, {}, {}};                  // full: sum over j<1024
  v8f accB[4] = {{}, {}, {}, {}};                  // same-class block chunk
  const int cb = qTile ? -1 : (t >> 1);            // block chunk index
  for (int ch = 0; ch < 32; ++ch) {                // j chunks of 32 (j<1024)
    const int j0 = ch * 32;
    const h8 lo = *(const h8*)&s_p[n][j0 + kbA];
    const h8 hi = *(const h8*)&s_p[n][j0 + kbA + 16];
    v16h a;
    #pragma unroll
    for (int e = 0; e < 8; ++e) { a[e] = lo[e]; a[8 + e] = hi[e]; }
    #pragma unroll
    for (int qq = 0; qq < 4; ++qq) {
      const int h0 = (wave + 8 * qq) * 16;
      const float* vp = Vg + (size_t)(b * S_ + j0 + kbB) * H_ + h0 + n;
      if (ch < 31) __builtin_prefetch(vp + (size_t)32 * H_, 0, 1);
      v16h bb;
      #pragma unroll
      for (int e = 0; e < 16; ++e) bb[e] = (_Float16)vp[(size_t)(e < 8 ? e : e + 8) * H_];
      accF[qq] = wmma_f16(a, bb, accF[qq]);
      if (ch == cb) accB[qq] = wmma_f16(a, bb, accB[qq]);
    }
  }

  // ---------------- epilogue: class-weight recombination ----------------
  #pragma unroll
  for (int qq = 0; qq < 4; ++qq) {
    const int h  = (wave + 8 * qq) * 16 + n;
    const float w0 = AW[0 * H_ + h], w1 = AW[1 * H_ + h], w2 = AW[2 * H_ + h];
    const float w3 = AW[3 * H_ + h], w4 = AW[4 * H_ + h], w5 = AW[5 * H_ + h];
    const float vNK = Vg[(size_t)(b * S_ + NK_) * H_ + h];
    #pragma unroll
    for (int r = 0; r < 8; ++r) {
      const int M = r + 8 * hhalf;
      const int i = 16 * t + M;
      if (!qTile) {
        const float d  = (float)s_p[M][16 * t + M];   // attn[i,i]
        const float s3 = (float)s_p[M][NK_];          // attn[i,NK]
        const float vi = Vg[(size_t)(b * S_ + i) * H_ + h];
        const float res = w2 * accF[qq][r] + (w1 - w2) * accB[qq][r]
                        + (w0 - w1) * d * vi + w3 * s3 * vNK;
        outg[(size_t)(b * S_ + i) * H_ + h] = res;
      } else if (M == 0) {                            // query row i == NK
        const float s5 = (float)s_p[0][NK_];          // attn[NK,NK]
        const float res = w4 * accF[qq][r] + w5 * s5 * vNK;
        outg[(size_t)(b * S_ + NK_) * H_ + h] = res;
      }
    }
  }
}

extern "C" void kernel_launch(void* const* d_in, const int* in_sizes, int n_in,
                              void* d_out, int out_size, void* d_ws, size_t ws_size,
                              hipStream_t stream) {
  const float* q  = (const float*)d_in[0];
  const float* k  = (const float*)d_in[1];
  const float* v  = (const float*)d_in[2];
  const float* aw = (const float*)d_in[3];
  // d_in[4], d_in[5] are N, K scalars (compile-time constants here).
  float* out  = (float*)d_out;
  float* attn = out + (size_t)B_ * S_ * H_;   // outputs: (out, attn) flat
  dim3 grid(JT_, B_);
  VWA_attn_fused_kernel<<<grid, dim3(256), 0, stream>>>(q, k, v, aw, out, attn);
}
